// SparseResBlock_28664611734293
// MI455X (gfx1250) — matrix operand order repaired and checked by reading, not compiled, not run on hardware
//
#include <hip/hip_runtime.h>

typedef __attribute__((ext_vector_type(16))) _Float16 v16h;
typedef __attribute__((ext_vector_type(8)))  _Float16 v8h;
typedef __attribute__((ext_vector_type(8)))  float    v8f;

#define N_VOX 200000
#define KOFF  27
#define A_CH  64
#define B_CH  128
#define EPSV  1e-5f
#define NTILES (N_VOX / 16)      // 12500 exact
#define WAVES_PER_BLOCK 8
#define STAT_BLOCKS 256

// ---------------------------------------------------------------------------
// neighbor-table construction: nbr[i][k] = gidx[k][e] where sidx[k][e] == i
// (sidx entries are unique within a k); default = N_VOX -> zero dummy row
// ---------------------------------------------------------------------------
__global__ void nbr_init_kernel(int* __restrict__ nbr, int total) {
    for (int i = blockIdx.x * blockDim.x + threadIdx.x; i < total;
         i += gridDim.x * blockDim.x)
        nbr[i] = N_VOX;
}

__global__ void nbr_fill_kernel(const int* __restrict__ gidx,
                                const int* __restrict__ sidx,
                                int* __restrict__ nbr) {
    const int total = KOFF * N_VOX;
    for (int idx = blockIdx.x * blockDim.x + threadIdx.x; idx < total;
         idx += gridDim.x * blockDim.x) {
        int k = idx / N_VOX;
        int s = sidx[idx];
        if (s < N_VOX) nbr[s * KOFF + k] = gidx[idx];
    }
}

// ---------------------------------------------------------------------------
// BN statistics: two-stage deterministic reduction (coalesced reads)
// ---------------------------------------------------------------------------
__global__ void bn_partial_kernel(const float* __restrict__ x, int n, int C,
                                  float* __restrict__ psum,
                                  float* __restrict__ psq) {
    const int c = threadIdx.x;                 // blockDim.x == C
    float s = 0.f, s2 = 0.f;
    for (int row = blockIdx.x; row < n; row += gridDim.x) {
        float v = x[(long)row * C + c];
        s += v; s2 += v * v;
    }
    psum[blockIdx.x * C + c] = s;
    psq [blockIdx.x * C + c] = s2;
}

__global__ void bn_reduce_kernel(const float* __restrict__ psum,
                                 const float* __restrict__ psq, int C, int nblk,
                                 float* __restrict__ sum, float* __restrict__ sq) {
    const int c = threadIdx.x;                 // blockDim.x == C
    float s = 0.f, s2 = 0.f;
    for (int b = 0; b < nblk; ++b) { s += psum[b * C + c]; s2 += psq[b * C + c]; }
    sum[c] = s; sq[c] = s2;
}

// ---------------------------------------------------------------------------
// BN apply + convert to f16; row n (dummy) is zeroed. Optionally also emit
// the raw (un-normalized) features as f16 (for the 1x1 skip conv).
// ---------------------------------------------------------------------------
template <int C>
__global__ void bn_apply_cvt_kernel(const float* __restrict__ x,
                                    const float* __restrict__ sum,
                                    const float* __restrict__ sq,
                                    const float* __restrict__ gamma,
                                    const float* __restrict__ beta,
                                    _Float16* __restrict__ ynorm,
                                    _Float16* __restrict__ yraw) {
    const long total = (long)(N_VOX + 1) * C;
    for (long idx = (long)blockIdx.x * blockDim.x + threadIdx.x; idx < total;
         idx += (long)gridDim.x * blockDim.x) {
        int row = (int)(idx / C);
        int c   = (int)(idx - (long)row * C);
        if (row == N_VOX) {
            ynorm[idx] = (_Float16)0.0f;
            if (yraw) yraw[idx] = (_Float16)0.0f;
            continue;
        }
        float mean = sum[c] * (1.0f / N_VOX);
        float var  = sq[c]  * (1.0f / N_VOX) - mean * mean;
        float inv  = rsqrtf(var + EPSV);
        float v    = x[idx];
        ynorm[idx] = (_Float16)((v - mean) * inv * gamma[c] + beta[c]);
        if (yraw) yraw[idx] = (_Float16)v;
    }
}

// ---------------------------------------------------------------------------
// Pack weights into per-lane WMMA B-fragment layout:
//   wp[k][kstep][ntile][lane][e]  (f16)
//   column n = ntile*16 + (lane & 15)
//   K index  = kstep*32 + (e < 8 ? 8*(lane>>4) + e : 16 + 8*(lane>>4) + (e - 8))
// so the conv kernel reads one contiguous v16h (32 B) per lane per WMMA.
// ---------------------------------------------------------------------------
__global__ void pack_w_kernel(const float* __restrict__ w, _Float16* __restrict__ wp,
                              int nk, int cin) {
    const int ksteps = cin / 32;
    const int total  = nk * cin * 128;         // permutation of all elements
    for (int idx = blockIdx.x * blockDim.x + threadIdx.x; idx < total;
         idx += gridDim.x * blockDim.x) {
        int e    = idx & 15;
        int lane = (idx >> 4) & 31;
        int nt   = (idx >> 9) & 7;
        int rest = idx >> 12;
        int kstep = rest % ksteps;
        int k     = rest / ksteps;
        int n  = nt * 16 + (lane & 15);
        int kk = kstep * 32 + ((e < 8) ? 8 * (lane >> 4) + e : 16 + 8 * (lane >> 4) + (e - 8));
        wp[idx] = (_Float16)w[((long)k * cin + kk) * 128 + n];
    }
}

// ---------------------------------------------------------------------------
// Async (ASYNCcnt-tracked) copy of `bytes` from global to LDS, spread over
// all 256 threads; 16 B per lane per instruction (512 B per wave-op).
// ---------------------------------------------------------------------------
template <int BYTES>
__device__ __forceinline__ void async_stage(const _Float16* __restrict__ gsrc,
                                            _Float16* ldst) {
    unsigned lds_a = (unsigned)(uintptr_t)ldst + (unsigned)threadIdx.x * 16u;
    unsigned long long g_a =
        (unsigned long long)(uintptr_t)gsrc + (unsigned long long)threadIdx.x * 16ull;
    constexpr int ITERS = BYTES / (256 * 16);
#pragma unroll
    for (int i = 0; i < ITERS; ++i) {
        asm volatile("global_load_async_to_lds_b128 %0, %1, off"
                     :: "v"(lds_a), "v"(g_a) : "memory");
        lds_a += 256u * 16u;
        g_a   += 256ull * 16ull;
    }
}

__device__ __forceinline__ void wait_async_all() {
    asm volatile("s_wait_asynccnt 0x0" ::: "memory");
}

// ---------------------------------------------------------------------------
// Gather-GEMM sparse conv: each wave computes a 16-row x 128-col f32 tile.
// A fragments gathered per-voxel from f16 features (ISA 16-bit A layout),
// B fragments read from double-buffered LDS weights, prefetched one k ahead
// via async LDS-DMA and shared by all 8 waves. SKIP folds in the 1x1 conv.
// ---------------------------------------------------------------------------
template <int CIN, bool SKIP>
__global__ __launch_bounds__(256) void conv_wmma_kernel(
    const _Float16* __restrict__ feat,      // (N_VOX+1) x CIN, f16
    const _Float16* __restrict__ wpack,     // 27 x (CIN/32)*8*32*16, f16
    const int*      __restrict__ nbr,       // N_VOX x 27
    const _Float16* __restrict__ skipfeat,  // (N_VOX+1) x 64, f16 (SKIP only)
    const _Float16* __restrict__ wnin,      // 2*8*32*16, f16       (SKIP only)
    float*          __restrict__ out)       // N_VOX x 128, f32
{
    constexpr int KSTEPS = CIN / 32;
    constexpr int KSZ    = KSTEPS * 8 * 32 * 16;   // halves of weights per k
    __shared__ _Float16 ldsw[2][KSZ];              // double buffer
    __shared__ _Float16 ldsn[SKIP ? 2 * 8 * 32 * 16 : 8];

    const int lane  = threadIdx.x & 31;
    const int wv    = threadIdx.x >> 5;
    const int tile  = blockIdx.x * WAVES_PER_BLOCK + wv;
    const bool active = (tile < NTILES);            // wave-uniform
    const int m     = lane & 15;
    const int r     = active ? (tile * 16 + m) : 0; // gathered voxel row
    const int abase = (lane < 16) ? 0 : 8;          // A-frag K sub-offset

    v8f acc[8] = {};

    if constexpr (SKIP)
        async_stage<2 * 8 * 32 * 16 * 2>(wnin, ldsn);   // 16 KB, once
    async_stage<KSZ * 2>(wpack, &ldsw[0][0]);           // k = 0
    wait_async_all();
    __syncthreads();

    for (int k = 0; k < KOFF; ++k) {
        if (k + 1 < KOFF)                               // prefetch next offset
            async_stage<KSZ * 2>(wpack + (long)(k + 1) * KSZ, &ldsw[(k + 1) & 1][0]);

        if (active) {
            const _Float16* ldsb = &ldsw[k & 1][0];
            const int j = nbr[r * KOFF + k];            // neighbor (or dummy)
            const _Float16* rp = feat + (long)j * CIN;
#pragma unroll
            for (int ks = 0; ks < KSTEPS; ++ks) {
                v8h lo = *(const v8h*)(rp + ks * 32 + abase);
                v8h hi = *(const v8h*)(rp + ks * 32 + 16 + abase);
                v16h Af;
#pragma unroll
                for (int i = 0; i < 8; ++i) { Af[i] = lo[i]; Af[i + 8] = hi[i]; }
#pragma unroll
                for (int nt = 0; nt < 8; ++nt) {
                    v16h Bf = *(const v16h*)(ldsb + ((ks * 8 + nt) * 32 + lane) * 16);
                    acc[nt] = __builtin_amdgcn_wmma_f32_16x16x32_f16(
                        false, Af, false, Bf, (short)0, acc[nt], false, false);
                }
            }
        }
        wait_async_all();     // this wave's prefetch of k+1 landed
        __syncthreads();      // all waves done reading buf k & prefetching
    }

    if constexpr (SKIP) {
        if (active) {
            const _Float16* rp = skipfeat + (long)r * 64;  // identity gather
#pragma unroll
            for (int ks = 0; ks < 2; ++ks) {
                v8h lo = *(const v8h*)(rp + ks * 32 + abase);
                v8h hi = *(const v8h*)(rp + ks * 32 + 16 + abase);
                v16h Af;
#pragma unroll
                for (int i = 0; i < 8; ++i) { Af[i] = lo[i]; Af[i + 8] = hi[i]; }
#pragma unroll
                for (int nt = 0; nt < 8; ++nt) {
                    v16h Bf = *(const v16h*)(ldsn + ((ks * 8 + nt) * 32 + lane) * 16);
                    acc[nt] = __builtin_amdgcn_wmma_f32_16x16x32_f16(
                        false, Af, false, Bf, (short)0, acc[nt], false, false);
                }
            }
        }
    }

    if (active) {
        // C layout: lanes 0-15 hold N=lane, M=v; lanes 16-31 hold N=lane-16, M=8+v
        const int mb = (lane < 16) ? 0 : 8;
#pragma unroll
        for (int nt = 0; nt < 8; ++nt) {
            const int c = nt * 16 + m;
#pragma unroll
            for (int v = 0; v < 8; ++v)
                out[(long)(tile * 16 + mb + v) * 128 + c] = acc[nt][v];
        }
    }
}

// ---------------------------------------------------------------------------
extern "C" void kernel_launch(void* const* d_in, const int* in_sizes, int n_in,
                              void* d_out, int out_size, void* d_ws, size_t ws_size,
                              hipStream_t stream) {
    const float* feats = (const float*)d_in[0];   // [N,64]
    const float* w1    = (const float*)d_in[1];   // [27,64,128]
    const float* w2    = (const float*)d_in[2];   // [27,128,128]
    const float* wnin  = (const float*)d_in[3];   // [64,128]
    const float* g1    = (const float*)d_in[4];
    const float* b1    = (const float*)d_in[5];
    const float* g2    = (const float*)d_in[6];
    const float* b2    = (const float*)d_in[7];
    const int*   gidx  = (const int*)d_in[8];     // [27,N]
    const int*   sidx  = (const int*)d_in[9];     // [27,N]
    float* out = (float*)d_out;

    // ---- workspace carve-up (256 B aligned) ----
    char*  ws  = (char*)d_ws;
    size_t off = 0;
    auto take = [&](size_t bytes) -> void* {
        void* p = ws + off;
        off = (off + bytes + 255) & ~(size_t)255;
        return p;
    };
    float* bn1_sum = (float*)take(64 * 4);
    float* bn1_sq  = (float*)take(64 * 4);
    float* bn2_sum = (float*)take(128 * 4);
    float* bn2_sq  = (float*)take(128 * 4);
    float* psum    = (float*)take((size_t)STAT_BLOCKS * 128 * 4);
    float* psq     = (float*)take((size_t)STAT_BLOCKS * 128 * 4);
    int*   nbr     = (int*)take((size_t)N_VOX * KOFF * 4);
    _Float16* hn16 = (_Float16*)take((size_t)(N_VOX + 1) * A_CH * 2); // BN1(feats) f16
    _Float16* rw16 = (_Float16*)take((size_t)(N_VOX + 1) * A_CH * 2); // raw feats f16
    _Float16* w1p  = (_Float16*)take((size_t)KOFF * A_CH * B_CH * 2);
    _Float16* w2p  = (_Float16*)take((size_t)KOFF * B_CH * B_CH * 2);
    _Float16* wnp  = (_Float16*)take((size_t)A_CH * B_CH * 2);
    float*    h1   = (float*)take((size_t)N_VOX * B_CH * 4);          // conv1 out f32
    _Float16* h216 = (_Float16*)take((size_t)(N_VOX + 1) * B_CH * 2); // BN2(h1) f16

    // 1) neighbor table
    nbr_init_kernel<<<2048, 256, 0, stream>>>(nbr, N_VOX * KOFF);
    nbr_fill_kernel<<<2048, 256, 0, stream>>>(gidx, sidx, nbr);

    // 2) BN1 stats + apply + f16 convert (also raw feats f16 for skip)
    bn_partial_kernel<<<STAT_BLOCKS, A_CH, 0, stream>>>(feats, N_VOX, A_CH, psum, psq);
    bn_reduce_kernel<<<1, A_CH, 0, stream>>>(psum, psq, A_CH, STAT_BLOCKS, bn1_sum, bn1_sq);
    bn_apply_cvt_kernel<A_CH><<<4096, 256, 0, stream>>>(feats, bn1_sum, bn1_sq,
                                                        g1, b1, hn16, rw16);

    // 3) pack weights into WMMA fragment layout
    pack_w_kernel<<<1024, 256, 0, stream>>>(w1,   w1p, KOFF, A_CH);
    pack_w_kernel<<<2048, 256, 0, stream>>>(w2,   w2p, KOFF, B_CH);
    pack_w_kernel<<<64,   256, 0, stream>>>(wnin, wnp, 1,    A_CH);

    const int conv_blocks = (NTILES + WAVES_PER_BLOCK - 1) / WAVES_PER_BLOCK; // 1563

    // 4) conv1: h1 = gatherGEMM(hn16, w1)
    conv_wmma_kernel<A_CH, false><<<conv_blocks, 256, 0, stream>>>(
        hn16, w1p, nbr, nullptr, nullptr, h1);

    // 5) BN2 stats + apply + f16 convert
    bn_partial_kernel<<<STAT_BLOCKS, B_CH, 0, stream>>>(h1, N_VOX, B_CH, psum, psq);
    bn_reduce_kernel<<<1, B_CH, 0, stream>>>(psum, psq, B_CH, STAT_BLOCKS, bn2_sum, bn2_sq);
    bn_apply_cvt_kernel<B_CH><<<8192, 256, 0, stream>>>(h1, bn2_sum, bn2_sq,
                                                        g2, b2, h216, nullptr);

    // 6) conv2 + fused 1x1 skip: out = gatherGEMM(h216, w2) + rw16 @ w_nin
    conv_wmma_kernel<B_CH, true><<<conv_blocks, 256, 0, stream>>>(
        h216, w2p, nbr, rw16, wnp, out);

    (void)in_sizes; (void)n_in; (void)out_size; (void)ws_size;
}
